// GraphAttentionLayer_16441134809607
// MI455X (gfx1250) — matrix operand order repaired and checked by reading, not compiled, not run on hardware
//
#include <hip/hip_runtime.h>

typedef __attribute__((ext_vector_type(16))) _Float16 v16h;
typedef __attribute__((ext_vector_type(8)))  float    v8f;

constexpr int   kN         = 8192;
constexpr int   kInF       = 128;
constexpr int   kOutF      = 64;
constexpr int   kSplit     = 4;
constexpr int   kKPerSplit = kN / kSplit;   // 2048
constexpr float kAlpha     = 0.2f;

__device__ __forceinline__ float lrelu(float x) { return x > 0.0f ? x : kAlpha * x; }

// Order-preserving float<->uint encoding so max() can be done with atomicMax(u32).
__device__ __forceinline__ unsigned f32_key(float f) {
  unsigned u = __float_as_uint(f);
  return (u & 0x80000000u) ? ~u : (u | 0x80000000u);
}
__device__ __forceinline__ float key_f32(unsigned k) {
  unsigned u = (k & 0x80000000u) ? (k & 0x7fffffffu) : ~k;
  return __uint_as_float(u);
}

__global__ void gat_init(unsigned* maxKey) {
  if (threadIdx.x == 0 && blockIdx.x == 0) *maxKey = 0u;  // key-space -inf
}

// One block (64 threads) per node row: Wh row, WhT (f16 feature-major), Wh1/Wh2, max(Wh2).
__global__ __launch_bounds__(64) void gat_prep(const float* __restrict__ h,
                                               const float* __restrict__ W,
                                               const float* __restrict__ a,
                                               _Float16* __restrict__ WhT,
                                               float* __restrict__ Wh1,
                                               float* __restrict__ Wh2,
                                               unsigned* __restrict__ maxKey) {
  __shared__ float hrow[kInF];
  __shared__ float red1[64];
  __shared__ float red2[64];
  const int t   = threadIdx.x;
  const int row = blockIdx.x;
  hrow[t]      = h[(size_t)row * kInF + t];
  hrow[t + 64] = h[(size_t)row * kInF + t + 64];
  __syncthreads();
  float acc = 0.0f;
#pragma unroll 8
  for (int k = 0; k < kInF; ++k) acc = fmaf(hrow[k], W[k * kOutF + t], acc);
  WhT[(size_t)t * kN + row] = (_Float16)acc;   // feature-major for WMMA B loads
  red1[t] = acc * a[t];
  red2[t] = acc * a[kOutF + t];
  __syncthreads();
  for (int s = 32; s > 0; s >>= 1) {
    if (t < s) { red1[t] += red1[t + s]; red2[t] += red2[t + s]; }
    __syncthreads();
  }
  if (t == 0) {
    Wh1[row] = red1[0];
    Wh2[row] = red2[0];
    atomicMax(maxKey, f32_key(red2[0]));
  }
}

// Fused masked softmax-attention: each wave owns (16-row tile, K-split of 2048 cols).
// A = P (f16 probs, built in registers with K contiguous per lane half, k = 16*hi + e),
// B tiles 0..3 = WhT (f16, lane = output feature, 16 contiguous K halfs per lane),
// B tile 4 = constant in-register ones-column -> D column 0 = row sums (free via WMMA).
__global__ __launch_bounds__(128) void gat_attn(const int* __restrict__ adj,
                                                const float* __restrict__ Wh1,
                                                const float* __restrict__ Wh2,
                                                const _Float16* __restrict__ WhT,
                                                const unsigned* __restrict__ maxKey,
                                                float* __restrict__ outAcc,
                                                float* __restrict__ rowAcc) {
  const int lane  = threadIdx.x & 31;
  const int wave  = threadIdx.x >> 5;
  const int job   = blockIdx.x * 4 + wave;  // 0..2047
  const int tile  = job & 511;              // 512 row tiles of 16
  const int split = job >> 9;               // 0..3 K splits
  const int i0    = tile * 16;
  const int hi    = lane >> 4;              // lane half selects K sub-chunk
  const int m     = lane & 15;              // row within tile (A), feature/col (B/D)
  const int row   = i0 + m;

  const float maxT = key_f32(*maxKey);
  const float s    = Wh1[row];
  const float mi   = lrelu(s + maxT);       // fixed stabilizer: exp(score-mi) <= 1

  const int* adjRow = adj + (size_t)row * kN;
  const _Float16* bp0 = WhT + (size_t)(0 * 16 + m) * kN + hi * 16;
  const _Float16* bp1 = WhT + (size_t)(1 * 16 + m) * kN + hi * 16;
  const _Float16* bp2 = WhT + (size_t)(2 * 16 + m) * kN + hi * 16;
  const _Float16* bp3 = WhT + (size_t)(3 * 16 + m) * kN + hi * 16;

  // Constant B fragment: column N=0 is all ones -> D(:,0) = sum_j p.  No memory traffic.
  v16h bones;
#pragma unroll
  for (int e = 0; e < 16; ++e) bones[e] = (m == 0) ? (_Float16)1.0f : (_Float16)0.0f;

  v8f acc0 = {}, acc1 = {}, acc2 = {}, acc3 = {}, acc4 = {};

  const int kbeg = split * kKPerSplit;
  const int kend = kbeg + kKPerSplit;
  for (int k0 = kbeg; k0 < kend; k0 += 32) {
    // A-layout K chunks for this lane: [c0..c0+7] -> af[0..7], [c1..c1+7] -> af[8..15]
    const int c0 = k0 + hi * 16;
    const int c1 = c0 + 8;
    __builtin_prefetch(adjRow + c0 + 256, 0, 1);   // global_prefetch ~8 iters ahead
    const int4 ad0 = *(const int4*)(adjRow + c0);
    const int4 ad1 = *(const int4*)(adjRow + c0 + 4);
    const int4 ad2 = *(const int4*)(adjRow + c1);
    const int4 ad3 = *(const int4*)(adjRow + c1 + 4);
    const float4 t0 = *(const float4*)(Wh2 + c0);
    const float4 t1 = *(const float4*)(Wh2 + c0 + 4);
    const float4 t2 = *(const float4*)(Wh2 + c1);
    const float4 t3 = *(const float4*)(Wh2 + c1 + 4);

    v16h af;
    auto mk = [&](int ab, float tv) -> _Float16 {
      float e = __expf(lrelu(s + tv) - mi);
      return (_Float16)(ab > 0 ? e : 0.0f);
    };
    af[0]  = mk(ad0.x, t0.x); af[1]  = mk(ad0.y, t0.y);
    af[2]  = mk(ad0.z, t0.z); af[3]  = mk(ad0.w, t0.w);
    af[4]  = mk(ad1.x, t1.x); af[5]  = mk(ad1.y, t1.y);
    af[6]  = mk(ad1.z, t1.z); af[7]  = mk(ad1.w, t1.w);
    af[8]  = mk(ad2.x, t2.x); af[9]  = mk(ad2.y, t2.y);
    af[10] = mk(ad2.z, t2.z); af[11] = mk(ad2.w, t2.w);
    af[12] = mk(ad3.x, t3.x); af[13] = mk(ad3.y, t3.y);
    af[14] = mk(ad3.z, t3.z); af[15] = mk(ad3.w, t3.w);

    acc0 = __builtin_amdgcn_wmma_f32_16x16x32_f16(false, af, false,
             *(const v16h*)(bp0 + k0), (short)0, acc0, false, false);
    acc1 = __builtin_amdgcn_wmma_f32_16x16x32_f16(false, af, false,
             *(const v16h*)(bp1 + k0), (short)0, acc1, false, false);
    acc2 = __builtin_amdgcn_wmma_f32_16x16x32_f16(false, af, false,
             *(const v16h*)(bp2 + k0), (short)0, acc2, false, false);
    acc3 = __builtin_amdgcn_wmma_f32_16x16x32_f16(false, af, false,
             *(const v16h*)(bp3 + k0), (short)0, acc3, false, false);
    acc4 = __builtin_amdgcn_wmma_f32_16x16x32_f16(false, af, false,
             bones, (short)0, acc4, false, false);
  }

  // D layout: VGPR r holds row r (lanes 0-15) / r+8 (lanes 16-31), col = m (+16*t).
  float* oa = outAcc + (size_t)split * kN * kOutF;
  v8f acc[4] = {acc0, acc1, acc2, acc3};
#pragma unroll
  for (int t = 0; t < 4; ++t) {
#pragma unroll
    for (int r = 0; r < 8; ++r) {
      const int orow = i0 + hi * 8 + r;
      oa[(size_t)orow * kOutF + t * 16 + m] = acc[t][r];
    }
  }
  // Row sums live in D-tile-4 column 0 (lanes with m==0).
  if (m == 0) {
#pragma unroll
    for (int r = 0; r < 8; ++r) {
      rowAcc[(size_t)split * kN + i0 + hi * 8 + r] = acc4[r];
    }
  }
}

// Deterministic fixed-order split reduction + normalize + ELU.
__global__ __launch_bounds__(256) void gat_finalize(const float* __restrict__ outAcc,
                                                    const float* __restrict__ rowAcc,
                                                    float* __restrict__ out) {
  const int gid = blockIdx.x * 256 + threadIdx.x;  // over 8192*64
  const int i = gid >> 6;
  float v = 0.0f;
#pragma unroll
  for (int sp = 0; sp < kSplit; ++sp) v += outAcc[(size_t)sp * kN * kOutF + gid];
  float rs = 0.0f;
#pragma unroll
  for (int sp = 0; sp < kSplit; ++sp) rs += rowAcc[sp * kN + i];
  rs = fmaxf(rs, 1e-30f);
  const float x = v / rs;
  out[gid] = x > 0.0f ? x : (__expf(x) - 1.0f);
}

extern "C" void kernel_launch(void* const* d_in, const int* in_sizes, int n_in,
                              void* d_out, int out_size, void* d_ws, size_t ws_size,
                              hipStream_t stream) {
  (void)in_sizes; (void)n_in; (void)out_size; (void)ws_size;
  const float* h   = (const float*)d_in[0];
  const int*   adj = (const int*)d_in[1];
  const float* W   = (const float*)d_in[2];
  const float* a   = (const float*)d_in[3];
  float* out = (float*)d_out;

  // Workspace layout (~9.6 MB total)
  char* ws = (char*)d_ws;
  _Float16* WhT = (_Float16*)ws;                                   // 64*8192*2 = 1,048,576 B
  float* Wh1 = (float*)(ws + (size_t)kOutF * kN * sizeof(_Float16));
  float* Wh2 = Wh1 + kN;
  unsigned* maxKey = (unsigned*)(Wh2 + kN);
  float* outAcc = (float*)(ws + 1114368);                          // 256B-aligned
  float* rowAcc = outAcc + (size_t)kSplit * kN * kOutF;            // +8,388,608 B

  gat_init<<<dim3(1), dim3(1), 0, stream>>>(maxKey);
  gat_prep<<<dim3(kN), dim3(64), 0, stream>>>(h, W, a, WhT, Wh1, Wh2, maxKey);
  gat_attn<<<dim3(512), dim3(128), 0, stream>>>(adj, Wh1, Wh2, WhT, maxKey, outAcc, rowAcc);
  gat_finalize<<<dim3(kN * kOutF / 256), dim3(256), 0, stream>>>(outAcc, rowAcc, out);
}
